// SingleHead_74285754351641
// MI455X (gfx1250) — matrix operand order repaired and checked by reading, not compiled, not run on hardware
//
#include <hip/hip_runtime.h>
#include <hip/hip_bf16.h>
#include <math.h>

typedef __bf16 bf16_t;
typedef __attribute__((ext_vector_type(16))) __bf16 v16bf;
typedef __attribute__((ext_vector_type(8)))  __bf16 v8bf;
typedef __attribute__((ext_vector_type(4)))  __bf16 v4bf;
typedef __attribute__((ext_vector_type(8)))  float  v8f;

#define DIM     1024
#define SEQ     2048
#define NBATCH  4
#define NTOK    (NBATCH * SEQ)   // 8192

// ---------------------------------------------------------------------------
// WMMA helpers (CDNA5 wave32, 16x16x32 bf16 -> f32)
// ---------------------------------------------------------------------------
__device__ __forceinline__ v8f wmma_bf16(v16bf a, v16bf b, v8f c) {
  return __builtin_amdgcn_wmma_f32_16x16x32_bf16(false, a, false, b,
                                                 (short)0, c, false, false);
}

// A-operand fragment (16x32, row = lane%16). ISA 7.12.2 16-bit A layout:
// group g = lane/16: elem i<8 -> K = g*8+i ; elem i>=8 -> K = 16+g*8+(i-8).
__device__ __forceinline__ v16bf load_fragA(const bf16_t* rowk, int g) {
  v8bf lo = *(const v8bf*)(rowk + g * 8);
  v8bf hi = *(const v8bf*)(rowk + 16 + g * 8);
  v16bf f;
#pragma unroll
  for (int i = 0; i < 8; ++i) { f[i] = lo[i]; f[i + 8] = hi[i]; }
  return f;
}

// B-operand fragment (32x16, col = lane%16), same K order as A:
// group g = lane/16: elem i<8 -> K = g*8+i ; elem i>=8 -> K = 16+g*8+(i-8).
__device__ __forceinline__ v16bf load_fragB(const bf16_t* colk, int g) {
  v8bf lo = *(const v8bf*)(colk + g * 8);
  v8bf hi = *(const v8bf*)(colk + 16 + g * 8);
  v16bf f;
#pragma unroll
  for (int i = 0; i < 8; ++i) { f[i] = lo[i]; f[i + 8] = hi[i]; }
  return f;
}

// ---------------------------------------------------------------------------
// f32 -> bf16 conversion (4-wide)
// ---------------------------------------------------------------------------
__global__ void __launch_bounds__(256)
cvt_f32_bf16(const float* __restrict__ in, bf16_t* __restrict__ out, int n4) {
  int i = blockIdx.x * blockDim.x + threadIdx.x;
  if (i < n4) {
    float4 v = ((const float4*)in)[i];
    v4bf o;
    o[0] = (bf16_t)v.x; o[1] = (bf16_t)v.y;
    o[2] = (bf16_t)v.z; o[3] = (bf16_t)v.w;
    ((v4bf*)out)[i] = o;
  }
}

// ---------------------------------------------------------------------------
// QKV projection:  Y = Xbf @ W^T + bias   (M=8192, N=1024, K=1024)
// block = 256 thr = 8 waves -> 128x64 tile; wave -> 32x32 tile.
// K-loop is double-buffered through LDS with async global->LDS copies:
//   global_load_async_to_lds_b128  (3 async ops / wave / stage)
//   s_wait_asynccnt + barrier, then ds_load_b128 fragment reads.
// storeT=0: Y[token][n] (bf16).  storeT=1: Yt[b][d][s] (bf16, for V).
// ---------------------------------------------------------------------------
__global__ void __launch_bounds__(256)
qkv_gemm_wmma(const bf16_t* __restrict__ X, const bf16_t* __restrict__ W,
              const float* __restrict__ bias, bf16_t* __restrict__ Y,
              int storeT) {
  __shared__ bf16_t Xs[2][128 * 32];   // 2 x 8 KB
  __shared__ bf16_t Ws[2][64 * 32];    // 2 x 4 KB

  const int tid  = threadIdx.x;
  const int lane = tid & 31;
  const int wave = tid >> 5;
  const int g    = lane >> 4;
  const int ln   = lane & 15;
  const int m0 = blockIdx.x * 128;
  const int n0 = blockIdx.y * 64;
  const int wm = (wave & 3) * 32;   // wave row offset in block tile
  const int wn = (wave >> 2) * 32;  // wave col offset in block tile

  // Stage one 32-wide k-slab of X (128x32) and W (64x32) into LDS buffer b.
  // Each 16B chunk is one lane of one async instruction; every wave issues
  // exactly 3 async instructions per stage (2 for X, 1 for W).
  auto stage = [&](int kb, int buf) {
    uint32_t xs = (uint32_t)(uintptr_t)(&Xs[buf][0]);
    uint32_t ws = (uint32_t)(uintptr_t)(&Ws[buf][0]);
#pragma unroll
    for (int c = tid; c < 512; c += 256) {           // X: 512 chunks
      int row = c >> 2, part = c & 3;
      uint64_t ga = (uint64_t)(uintptr_t)(X + (size_t)(m0 + row) * DIM + kb + part * 8);
      uint32_t la = xs + (uint32_t)c * 16;
      asm volatile("global_load_async_to_lds_b128 %0, %1, off"
                   :: "v"(la), "v"(ga) : "memory");
    }
    {                                                 // W: 256 chunks
      int c = tid, row = c >> 2, part = c & 3;
      uint64_t ga = (uint64_t)(uintptr_t)(W + (size_t)(n0 + row) * DIM + kb + part * 8);
      uint32_t la = ws + (uint32_t)c * 16;
      asm volatile("global_load_async_to_lds_b128 %0, %1, off"
                   :: "v"(la), "v"(ga) : "memory");
    }
  };

  v8f c00 = {}, c01 = {}, c10 = {}, c11 = {};
  stage(0, 0);

  for (int kb = 0; kb < DIM; kb += 32) {
    const int cur = (kb >> 5) & 1;
    if (kb + 32 < DIM) {
      stage(kb + 32, cur ^ 1);                         // prefetch next slab
      asm volatile("s_wait_asynccnt 0x3" ::: "memory"); // our slab done
    } else {
      asm volatile("s_wait_asynccnt 0x0" ::: "memory");
    }
    __syncthreads();                                   // everyone's slab done

    v16bf a0 = load_fragA(&Xs[cur][(wm + ln) * 32], g);
    v16bf a1 = load_fragA(&Xs[cur][(wm + 16 + ln) * 32], g);
    v16bf b0 = load_fragB(&Ws[cur][(wn + ln) * 32], g);
    v16bf b1 = load_fragB(&Ws[cur][(wn + 16 + ln) * 32], g);
    c00 = wmma_bf16(a0, b0, c00);
    c01 = wmma_bf16(a0, b1, c01);
    c10 = wmma_bf16(a1, b0, c10);
    c11 = wmma_bf16(a1, b1, c11);

    __syncthreads();                                   // done reading cur slab
  }

  const float bn0 = bias[n0 + wn + ln];
  const float bn1 = bias[n0 + wn + 16 + ln];
#pragma unroll
  for (int r = 0; r < 8; ++r) {
    int ma = m0 + wm + r + 8 * g;
    int mb = ma + 16;
    int na = n0 + wn + ln;
    int nb = na + 16;
    float v00 = c00[r] + bn0, v01 = c01[r] + bn1;
    float v10 = c10[r] + bn0, v11 = c11[r] + bn1;
    if (!storeT) {
      Y[(size_t)ma * DIM + na] = (bf16_t)v00;
      Y[(size_t)ma * DIM + nb] = (bf16_t)v01;
      Y[(size_t)mb * DIM + na] = (bf16_t)v10;
      Y[(size_t)mb * DIM + nb] = (bf16_t)v11;
    } else {  // Vt[b][d][s]
      size_t ba = (size_t)(ma >> 11) * DIM * SEQ, sa = (size_t)(ma & 2047);
      size_t bb = (size_t)(mb >> 11) * DIM * SEQ, sb = (size_t)(mb & 2047);
      Y[ba + (size_t)na * SEQ + sa] = (bf16_t)v00;
      Y[ba + (size_t)nb * SEQ + sa] = (bf16_t)v01;
      Y[bb + (size_t)na * SEQ + sb] = (bf16_t)v10;
      Y[bb + (size_t)nb * SEQ + sb] = (bf16_t)v11;
    }
  }
}

// ---------------------------------------------------------------------------
// Flash attention (causal). 4 waves / block, each wave owns 16 query rows.
// O accumulator (16x1024 f32 / wave) lives in LDS (4*64KB = 256KB < 320KB).
// P (16x32) round-trips through LDS to convert C-layout -> A-layout.
// ---------------------------------------------------------------------------
__global__ void __launch_bounds__(128)
attn_flash_wmma(const bf16_t* __restrict__ Qb, const bf16_t* __restrict__ Kb,
                const bf16_t* __restrict__ Vt, float* __restrict__ Out) {
  extern __shared__ char smem[];
  const int lane = threadIdx.x & 31;
  const int wave = threadIdx.x >> 5;
  const int g    = lane >> 4;
  const int ln   = lane & 15;
  const int b    = blockIdx.y;
  const int q0   = blockIdx.x * 64 + wave * 16;

  float*  Ow = (float*)smem + (size_t)wave * 16 * DIM;
  bf16_t* Pw = (bf16_t*)((float*)smem + 4 * 16 * DIM) + wave * (16 * 32);

  // zero O accumulator (per-wave region)
  float4 z4 = {0.f, 0.f, 0.f, 0.f};
  for (int i = lane; i < 16 * DIM / 4; i += 32) ((float4*)Ow)[i] = z4;

  float mrow[8], lsum[8];
#pragma unroll
  for (int r = 0; r < 8; ++r) { mrow[r] = -1e30f; lsum[r] = 0.f; }

  const bf16_t* Qrow  = Qb + ((size_t)b * SEQ + q0 + ln) * DIM;
  const bf16_t* Kbat  = Kb + (size_t)b * SEQ * DIM;
  const bf16_t* Vbat  = Vt + (size_t)b * DIM * SEQ;
  const float   scale = 0.03125f;            // 1/sqrt(1024)
  const int     nkt   = (q0 + 16 + 31) >> 5; // 32-key tiles covering causal span

  for (int kt = 0; kt < nkt; ++kt) {
    // ---- scores: 16x32 = two 16x16 wmma accumulations over d=1024 ----
    int kr0 = kt * 32 + ln, kr1 = kr0 + 16;
    const bf16_t* Kr0 = Kbat + (size_t)(kr0 < SEQ ? kr0 : SEQ - 1) * DIM;
    const bf16_t* Kr1 = Kbat + (size_t)(kr1 < SEQ ? kr1 : SEQ - 1) * DIM;
    if (kt + 1 < nkt) {  // prefetch next key tile (global_prefetch_b8)
      int nk = kr0 + 32;
      __builtin_prefetch(Kbat + (size_t)(nk < SEQ ? nk : SEQ - 1) * DIM, 0, 1);
    }
    v8f sc0 = {}, sc1 = {};
#pragma unroll 2
    for (int k = 0; k < DIM; k += 32) {
      v16bf aq = load_fragA(Qrow + k, g);
      sc0 = wmma_bf16(aq, load_fragB(Kr0 + k, g), sc0);
      sc1 = wmma_bf16(aq, load_fragB(Kr1 + k, g), sc1);
    }

    // ---- online softmax (row m = r+8g lives in lanes of group g) ----
    const int j0 = kt * 32 + ln, j1 = j0 + 16;
    float alpha[8];
#pragma unroll
    for (int r = 0; r < 8; ++r) {
      int q  = q0 + r + 8 * g;
      float v0 = (j0 <= q) ? sc0[r] * scale : -1e30f;
      float v1 = (j1 <= q) ? sc1[r] * scale : -1e30f;
      float tmax = fmaxf(v0, v1);
#pragma unroll
      for (int m = 8; m >= 1; m >>= 1) tmax = fmaxf(tmax, __shfl_xor(tmax, m, 16));
      float nm = fmaxf(mrow[r], tmax);
      float a_ = __expf(mrow[r] - nm);
      float e0 = __expf(v0 - nm);
      float e1 = __expf(v1 - nm);
      float rs = e0 + e1;
#pragma unroll
      for (int m = 8; m >= 1; m >>= 1) rs += __shfl_xor(rs, m, 16);
      lsum[r]  = lsum[r] * a_ + rs;
      mrow[r]  = nm;
      alpha[r] = a_;
      // stage P tile (C-layout) into LDS, bf16
      int mrowi = r + 8 * g;
      Pw[mrowi * 32 + ln]      = (bf16_t)e0;
      Pw[mrowi * 32 + 16 + ln] = (bf16_t)e1;
    }
    asm volatile("s_wait_dscnt 0" ::: "memory");

    // reload P as an A-operand fragment (same for every d-chunk)
    v16bf ap = load_fragA(Pw + ln * 32, g);

    // ---- O = alpha*O + P @ V  over 64 d-chunks of 16 cols ----
    const bf16_t* Vk = Vbat + kt * 32;
    for (int c = 0; c < DIM / 16; ++c) {
      v16bf bv = load_fragB(Vk + (size_t)(c * 16 + ln) * SEQ, g);
      v8f o;
#pragma unroll
      for (int r = 0; r < 8; ++r)
        o[r] = Ow[(r + 8 * g) * DIM + c * 16 + ln] * alpha[r];
      o = wmma_bf16(ap, bv, o);
#pragma unroll
      for (int r = 0; r < 8; ++r)
        Ow[(r + 8 * g) * DIM + c * 16 + ln] = o[r];
    }
  }

  // ---- normalize + write out (f32) ----
  float inv[8];
#pragma unroll
  for (int r = 0; r < 8; ++r) inv[r] = 1.f / lsum[r];
  for (int c = 0; c < DIM / 16; ++c) {
#pragma unroll
    for (int r = 0; r < 8; ++r) {
      int m = q0 + r + 8 * g;
      Out[((size_t)b * SEQ + m) * DIM + c * 16 + ln] =
          Ow[(r + 8 * g) * DIM + c * 16 + ln] * inv[r];
    }
  }
}

// ---------------------------------------------------------------------------
// Launch
// ---------------------------------------------------------------------------
extern "C" void kernel_launch(void* const* d_in, const int* in_sizes, int n_in,
                              void* d_out, int out_size, void* d_ws, size_t ws_size,
                              hipStream_t stream) {
  const float* x  = (const float*)d_in[0];
  const float* Wk = (const float*)d_in[1];
  const float* bk = (const float*)d_in[2];
  const float* Wq = (const float*)d_in[3];
  const float* bq = (const float*)d_in[4];
  const float* Wv = (const float*)d_in[5];
  const float* bv = (const float*)d_in[6];
  float* out = (float*)d_out;

  // workspace layout (bf16): xb | Wqb | Wkb | Wvb | Qb | Kb | Vtb  (~70 MB)
  bf16_t* xb  = (bf16_t*)d_ws;
  bf16_t* Wqb = xb  + (size_t)NTOK * DIM;
  bf16_t* Wkb = Wqb + (size_t)DIM * DIM;
  bf16_t* Wvb = Wkb + (size_t)DIM * DIM;
  bf16_t* Qb  = Wvb + (size_t)DIM * DIM;
  bf16_t* Kb  = Qb  + (size_t)NTOK * DIM;
  bf16_t* Vtb = Kb  + (size_t)NTOK * DIM;

  // 1) convert inputs to bf16
  int n4x = NTOK * DIM / 4;
  cvt_f32_bf16<<<(n4x + 255) / 256, 256, 0, stream>>>(x, xb, n4x);
  int n4w = DIM * DIM / 4;
  cvt_f32_bf16<<<(n4w + 255) / 256, 256, 0, stream>>>(Wq, Wqb, n4w);
  cvt_f32_bf16<<<(n4w + 255) / 256, 256, 0, stream>>>(Wk, Wkb, n4w);
  cvt_f32_bf16<<<(n4w + 255) / 256, 256, 0, stream>>>(Wv, Wvb, n4w);

  // 2) QKV projections (V stored d-major transposed)
  dim3 gg(NTOK / 128, DIM / 64), bb(256);
  qkv_gemm_wmma<<<gg, bb, 0, stream>>>(xb, Wqb, bq, Qb, 0);
  qkv_gemm_wmma<<<gg, bb, 0, stream>>>(xb, Wkb, bk, Kb, 0);
  qkv_gemm_wmma<<<gg, bb, 0, stream>>>(xb, Wvb, bv, Vtb, 1);

  // 3) causal flash attention
  size_t shmem = (size_t)4 * 16 * DIM * sizeof(float)     // O: 256 KB
               + (size_t)4 * 16 * 32 * sizeof(bf16_t);    // P:   4 KB
  (void)hipFuncSetAttribute(reinterpret_cast<const void*>(attn_flash_wmma),
                            hipFuncAttributeMaxDynamicSharedMemorySize,
                            (int)shmem);
  attn_flash_wmma<<<dim3(SEQ / 64, NBATCH), 128, shmem, stream>>>(Qb, Kb, Vtb, out);
}
